// Block_58377195487260
// MI455X (gfx1250) — matrix operand order repaired and checked by reading, not compiled, not run on hardware
//
#include <hip/hip_runtime.h>
#include <hip/hip_bf16.h>
#include <math.h>

typedef __attribute__((ext_vector_type(16))) _Float16 v16h;
typedef __attribute__((ext_vector_type(8)))  _Float16 v8h;
typedef __attribute__((ext_vector_type(8)))  float    v8f;

#define DEV __device__ __forceinline__

DEV v8f zero8() { v8f z = {0.f,0.f,0.f,0.f,0.f,0.f,0.f,0.f}; return z; }

// ---- CDNA5 WMMA fragment loads (wave32, v_wmma_f32_16x16x32_f16) ----
// A (16x32 f16, MxK): lanes 0-15 -> M=lane, K 0..15;
//                     lanes 16-31 -> M=lane-16, K 16..31.
// Per lane: two contiguous 16B chunks of row M.
DEV v16h load_a_frag(const _Float16* __restrict__ A, int lda, int lane) {
  const int row = lane & 15;
  const int kb  = (lane >> 4) << 4;           // 0 or 16
  const _Float16* p = A + (size_t)row * lda + kb;
  v8h lo = *(const v8h*)(p);
  v8h hi = *(const v8h*)(p + 8);
  return __builtin_shufflevector(lo, hi, 0,1,2,3,4,5,6,7,8,9,10,11,12,13,14,15);
}

// B (32x16 f16, KxN), provided TRANSPOSED in memory as Bt[N][K]:
// lane holds column N=lane&15, K = (lane<16 ? 0..15 : 16..31): contiguous 32B.
DEV v16h load_b_frag(const _Float16* __restrict__ Bt, int ldb, int lane) {
  const int col = lane & 15;
  const int kb  = (lane >> 4) << 4;           // 0 or 16
  return *(const v16h*)(Bt + (size_t)col * ldb + kb);
}

DEV v8f wmma16(v16h a, v16h b, v8f c) {
  return __builtin_amdgcn_wmma_f32_16x16x32_f16(false, a, false, b, (short)0, c, false, false);
}

DEV float gelu_exact(float x) { return 0.5f * x * (1.0f + erff(x * 0.70710678118654752f)); }

// ---------------- transpose + convert to f16: out[c][r] = in[r][c] ------------
template <typename Tin>
__global__ __launch_bounds__(256)
void transpose_to_f16(const Tin* __restrict__ in, _Float16* __restrict__ out, int R, int C) {
  __shared__ float tile[32][33];
  const Tin* inb = in + (size_t)blockIdx.z * R * C;
  _Float16* outb = out + (size_t)blockIdx.z * R * C;
  int c = blockIdx.x * 32 + threadIdx.x;
  #pragma unroll
  for (int j = 0; j < 32; j += 8) {
    int r = blockIdx.y * 32 + threadIdx.y + j;
    tile[threadIdx.y + j][threadIdx.x] = (float)inb[(size_t)r * C + c];
  }
  __syncthreads();
  int rr = blockIdx.y * 32 + threadIdx.x;
  #pragma unroll
  for (int j = 0; j < 32; j += 8) {
    int cc = blockIdx.x * 32 + threadIdx.y + j;
    outb[(size_t)cc * R + rr] = (_Float16)tile[threadIdx.x][threadIdx.y + j];
  }
}

// ---------------- LayerNorm over D=1024, one row per block -------------------
__global__ __launch_bounds__(256)
void ln_kernel(const float* __restrict__ x, const float* __restrict__ g,
               const float* __restrict__ b, float* __restrict__ out32,
               _Float16* __restrict__ out16, int D) {
  __shared__ float rs[256], rq[256];
  const int tid = threadIdx.x;
  const size_t base = (size_t)blockIdx.x * D;
  float s = 0.f, q = 0.f;
  float v[4];
  #pragma unroll
  for (int t = 0; t < 4; ++t) {
    v[t] = x[base + t * 256 + tid];
    s += v[t]; q += v[t] * v[t];
  }
  rs[tid] = s; rq[tid] = q;
  __syncthreads();
  for (int o = 128; o; o >>= 1) {
    if (tid < o) { rs[tid] += rs[tid + o]; rq[tid] += rq[tid + o]; }
    __syncthreads();
  }
  const float mu = rs[0] / (float)D;
  const float var = rq[0] / (float)D - mu * mu;
  const float rstd = rsqrtf(var + 1e-5f);
  #pragma unroll
  for (int t = 0; t < 4; ++t) {
    int i = t * 256 + tid;
    float o = (v[t] - mu) * rstd * g[i] + b[i];
    if (out32) out32[base + i] = o;
    out16[base + i] = (_Float16)o;
  }
}

// ---------------- generic WMMA GEMM: C[M,N] = A[M,K] @ Bt[N,K]^T -------------
// EPI bits: 1=bias[n], 2=gelu, 4=add residual f32, 8=store f16 (else f32)
template <int EPI>
__global__ __launch_bounds__(256)
void gemm_wmma(const _Float16* __restrict__ A, const _Float16* __restrict__ Bt,
               float* __restrict__ C32, _Float16* __restrict__ C16,
               const float* __restrict__ bias, const float* __restrict__ res,
               int M, int N, int K) {
  const int lane = threadIdx.x & 31;
  const int w = threadIdx.x >> 5;
  const int m0 = blockIdx.y * 128 + w * 16;
  const int n0 = blockIdx.x * 128;

  v8f acc[8];
  #pragma unroll
  for (int t = 0; t < 8; ++t) acc[t] = zero8();

  for (int k0 = 0; k0 < K; k0 += 32) {
    v16h a = load_a_frag(A + (size_t)m0 * K + k0, K, lane);
    #pragma unroll
    for (int t = 0; t < 8; ++t) {
      v16h b = load_b_frag(Bt + (size_t)(n0 + 16 * t) * K + k0, K, lane);
      acc[t] = wmma16(a, b, acc[t]);
    }
  }

  const int nl = lane & 15;
  const int mb = m0 + ((lane >> 4) << 3);
  #pragma unroll
  for (int t = 0; t < 8; ++t) {
    const int n = n0 + 16 * t + nl;
    const float bv = (EPI & 1) ? bias[n] : 0.0f;
    #pragma unroll
    for (int r = 0; r < 8; ++r) {
      float v = acc[t][r] + bv;
      if (EPI & 2) v = gelu_exact(v);
      const size_t idx = (size_t)(mb + r) * N + n;
      if (EPI & 4) v += res[idx];
      if (EPI & 8) C16[idx] = (_Float16)v;
      else         C32[idx] = v;
    }
  }
}

// ---------------- attention: one block per (batch, 16-query tile) ------------
// Q,K: [B,S,D] f16 row-major; Vt: [B,D,S] f16 (pre-transposed); Y: [B,S,D] f16.
// S=2048, D=1024 hardcoded. LDS: 16x2048 f16 scores/probs, XOR-swizzled 16B groups.
DEV int sw_idx(int row, int k) {
  return row * 2048 + ((((k >> 3) ^ row) & 255) << 3) + (k & 7);
}

__global__ __launch_bounds__(256)
void attn_kernel(const _Float16* __restrict__ Q, const _Float16* __restrict__ Kt,
                 const _Float16* __restrict__ Vt, _Float16* __restrict__ Y) {
  constexpr int S = 2048, D = 1024;
  __shared__ _Float16 Sbuf[16 * 2048];    // 64 KB
  const int lane = threadIdx.x & 31;
  const int w = threadIdx.x >> 5;
  const int b = blockIdx.y;
  const int q0 = blockIdx.x * 16;
  const _Float16* Qb = Q + ((size_t)b * S + q0) * D;
  const _Float16* Kb = Kt + (size_t)b * S * D;
  const _Float16* Vb = Vt + (size_t)b * D * S;
  const float scale = 0.03125f;           // 1/sqrt(1024)

  // Phase 1: scaled scores S[q, key] for keys [w*256, w*256+256)
  for (int kt = 0; kt < 16; ++kt) {
    const int key0 = w * 256 + kt * 16;
    v8f acc = zero8();
    for (int d0 = 0; d0 < D; d0 += 32) {
      v16h a = load_a_frag(Qb + d0, D, lane);
      v16h bb = load_b_frag(Kb + (size_t)key0 * D + d0, D, lane);  // K row-major == Kt^T column-contiguous
      acc = wmma16(a, bb, acc);
    }
    const int col = key0 + (lane & 15);
    const int mb = (lane >> 4) << 3;
    #pragma unroll
    for (int r = 0; r < 8; ++r)
      Sbuf[sw_idx(mb + r, col)] = (_Float16)(acc[r] * scale);
  }
  __syncthreads();

  // Phase 2: row softmax, 16 threads per row, 128 elements each, in place
  {
    const int row = threadIdx.x >> 4;
    const int c = threadIdx.x & 15;
    float mx = -1e30f;
    for (int i = 0; i < 128; ++i)
      mx = fmaxf(mx, (float)Sbuf[sw_idx(row, c * 128 + i)]);
    for (int o = 8; o; o >>= 1) mx = fmaxf(mx, __shfl_xor(mx, o, 32));
    float sum = 0.f;
    for (int i = 0; i < 128; ++i)
      sum += __expf((float)Sbuf[sw_idx(row, c * 128 + i)] - mx);
    for (int o = 8; o; o >>= 1) sum += __shfl_xor(sum, o, 32);
    const float inv = 1.0f / sum;
    for (int i = 0; i < 128; ++i) {
      const int ii = sw_idx(row, c * 128 + i);
      Sbuf[ii] = (_Float16)(__expf((float)Sbuf[ii] - mx) * inv);
    }
  }
  __syncthreads();

  // Phase 3: Y[q, n] = P @ V, wave handles 128 value-columns
  const int n0w = w * 128;
  v8f acc[8];
  #pragma unroll
  for (int t = 0; t < 8; ++t) acc[t] = zero8();
  for (int k0 = 0; k0 < S; k0 += 32) {
    const int row = lane & 15;
    const int kb = k0 + ((lane >> 4) << 4);
    v8h lo = *(const v8h*)(Sbuf + sw_idx(row, kb));
    v8h hi = *(const v8h*)(Sbuf + sw_idx(row, kb + 8));
    v16h a = __builtin_shufflevector(lo, hi, 0,1,2,3,4,5,6,7,8,9,10,11,12,13,14,15);
    #pragma unroll
    for (int t = 0; t < 8; ++t) {
      v16h bb = load_b_frag(Vb + (size_t)(n0w + 16 * t) * S + k0, S, lane);
      acc[t] = wmma16(a, bb, acc[t]);
    }
  }
  const int nl = lane & 15;
  const int mb2 = (lane >> 4) << 3;
  #pragma unroll
  for (int t = 0; t < 8; ++t)
    #pragma unroll
    for (int r = 0; r < 8; ++r)
      Y[((size_t)b * S + q0 + mb2 + r) * D + n0w + 16 * t + nl] = (_Float16)acc[t][r];
}

// -----------------------------------------------------------------------------
extern "C" void kernel_launch(void* const* d_in, const int* in_sizes, int n_in,
                              void* d_out, int out_size, void* d_ws, size_t ws_size,
                              hipStream_t stream) {
  (void)in_sizes; (void)n_in; (void)out_size; (void)ws_size;
  constexpr int B = 8, S = 2048, D = 1024, H = 4096;
  constexpr int M = B * S;

  const float* x     = (const float*)d_in[0];
  const float* ln1_g = (const float*)d_in[1];
  const float* ln1_b = (const float*)d_in[2];
  const float* wq    = (const float*)d_in[3];
  const float* wk    = (const float*)d_in[4];
  const float* wv    = (const float*)d_in[5];
  const float* wo    = (const float*)d_in[6];
  const float* ln2_g = (const float*)d_in[7];
  const float* ln2_b = (const float*)d_in[8];
  const float* w1    = (const float*)d_in[9];
  const float* b1    = (const float*)d_in[10];
  const float* w2    = (const float*)d_in[11];
  const float* b2    = (const float*)d_in[12];
  float* out = (float*)d_out;

  size_t off = 0;
  auto take = [&](size_t bytes) -> char* {
    char* p = (char*)d_ws + off;
    off += (bytes + 255) & ~(size_t)255;
    return p;
  };
  float*    h32  = (float*)   take((size_t)M * D * 4);
  _Float16* h16  = (_Float16*)take((size_t)M * D * 2);
  _Float16* q16  = (_Float16*)take((size_t)M * D * 2);
  _Float16* k16  = (_Float16*)take((size_t)M * D * 2);
  _Float16* vv16 = (_Float16*)take((size_t)M * D * 2);
  _Float16* vt16 = (_Float16*)take((size_t)M * D * 2);
  _Float16* y16  = (_Float16*)take((size_t)M * D * 2);
  float*    x2   = (float*)   take((size_t)M * D * 4);
  _Float16* h2   = (_Float16*)take((size_t)M * D * 2);
  _Float16* u16  = (_Float16*)take((size_t)M * H * 2);
  _Float16* wqT  = (_Float16*)take((size_t)D * D * 2);
  _Float16* wkT  = (_Float16*)take((size_t)D * D * 2);
  _Float16* wvT  = (_Float16*)take((size_t)D * D * 2);
  _Float16* woT  = (_Float16*)take((size_t)D * D * 2);
  _Float16* w1T  = (_Float16*)take((size_t)D * H * 2);
  _Float16* w2T  = (_Float16*)take((size_t)H * D * 2);

  const dim3 tb(32, 8);
  // weight transposes (f32 -> f16, [K,N] -> [N,K])
  transpose_to_f16<float><<<dim3(D / 32, D / 32, 1), tb, 0, stream>>>(wq, wqT, D, D);
  transpose_to_f16<float><<<dim3(D / 32, D / 32, 1), tb, 0, stream>>>(wk, wkT, D, D);
  transpose_to_f16<float><<<dim3(D / 32, D / 32, 1), tb, 0, stream>>>(wv, wvT, D, D);
  transpose_to_f16<float><<<dim3(D / 32, D / 32, 1), tb, 0, stream>>>(wo, woT, D, D);
  transpose_to_f16<float><<<dim3(H / 32, D / 32, 1), tb, 0, stream>>>(w1, w1T, D, H);
  transpose_to_f16<float><<<dim3(D / 32, H / 32, 1), tb, 0, stream>>>(w2, w2T, H, D);

  // h = LN1(x), kept in f32 (residual) and f16 (GEMM input)
  ln_kernel<<<M, 256, 0, stream>>>(x, ln1_g, ln1_b, h32, h16, D);

  // q, k, v projections (f16 outputs)
  gemm_wmma<8><<<dim3(D / 128, M / 128), 256, 0, stream>>>(h16, wqT, nullptr, q16, nullptr, nullptr, M, D, D);
  gemm_wmma<8><<<dim3(D / 128, M / 128), 256, 0, stream>>>(h16, wkT, nullptr, k16, nullptr, nullptr, M, D, D);
  gemm_wmma<8><<<dim3(D / 128, M / 128), 256, 0, stream>>>(h16, wvT, nullptr, vv16, nullptr, nullptr, M, D, D);

  // V -> Vt per batch ([S,D] -> [D,S])
  transpose_to_f16<_Float16><<<dim3(D / 32, S / 32, B), tb, 0, stream>>>(vv16, vt16, S, D);

  // attention
  attn_kernel<<<dim3(S / 16, B), 256, 0, stream>>>(q16, k16, vt16, y16);

  // x2 = y @ wo + h  (f32)
  gemm_wmma<4><<<dim3(D / 128, M / 128), 256, 0, stream>>>(y16, woT, x2, nullptr, nullptr, h32, M, D, D);

  // h2 = LN2(x2) (f16 only)
  ln_kernel<<<M, 256, 0, stream>>>(x2, ln2_g, ln2_b, nullptr, h2, D);

  // u = gelu(h2 @ w1 + b1) (f16)
  gemm_wmma<11><<<dim3(H / 128, M / 128), 256, 0, stream>>>(h2, w1T, nullptr, u16, b1, nullptr, M, H, D);

  // out = u @ w2 + b2 + x2 (f32)
  gemm_wmma<5><<<dim3(D / 128, M / 128), 256, 0, stream>>>(u16, w2T, out, nullptr, b2, x2, M, D, H);
}
